// MoEMlp_56384330662505
// MI455X (gfx1250) — matrix operand order repaired and checked by reading, not compiled, not run on hardware
//
#include <hip/hip_runtime.h>
#include <math.h>
#include <stdint.h>

// ---------------- problem constants (from reference setup_inputs) ----------
#define NTOK   16384       // B*S = 4*4096
#define DIM    1024
#define HID    4096
#define NEXP   8
#define TOPK   2
#define CAP    5120        // ceil(TOPK*N/E*1.25)
#define NA     (NTOK*TOPK) // 32768 assignments
#define HC     512         // H chunk size
#define NCHUNK (HID/HC)    // 8

typedef __attribute__((ext_vector_type(16))) __bf16 v16bf;
typedef __attribute__((ext_vector_type(8)))  float  v8f;

// async 16B copy global -> LDS (gfx1250, ASYNCcnt). lds = byte offset in LDS.
__device__ __forceinline__ void async_b128(unsigned lds, const __bf16* src) {
    asm volatile("global_load_async_to_lds_b128 %0, %1, off"
                 :: "v"(lds), "v"(src) : "memory");
}
__device__ __forceinline__ void wait_async0() {
    asm volatile("s_wait_asynccnt 0x0" ::: "memory");
}
// ASYNCcnt completes in order: <=4 means everything older than the 4 newest is done.
__device__ __forceinline__ void wait_async4() {
    asm volatile("s_wait_asynccnt 0x4" ::: "memory");
}

// ---------------------------------------------------------------- init
__global__ void init_kernel(int* slot_token, __bf16* zrow) {
    int i = blockIdx.x * blockDim.x + threadIdx.x;
    if (i < NEXP * CAP) slot_token[i] = -1;
    if (i < DIM) zrow[i] = (__bf16)0.f;
}

// ------------------------------------------------------- x: f32 -> bf16 copy
__global__ void cvt_x_kernel(const float* __restrict__ x, __bf16* __restrict__ xb) {
    size_t i = ((size_t)blockIdx.x * 256 + threadIdx.x) * 4;
    float4 v = *(const float4*)(x + i);
    xb[i + 0] = (__bf16)v.x; xb[i + 1] = (__bf16)v.y;
    xb[i + 2] = (__bf16)v.z; xb[i + 3] = (__bf16)v.w;
}

// ---------------------------- weights: f32 [E][R][C] -> bf16 [E][C][R] (transpose)
__global__ void transpose_cvt_kernel(const float* __restrict__ in,
                                     __bf16* __restrict__ out, int R, int C) {
    __shared__ float t[32][33];
    int e = blockIdx.z;
    const float* ine = in + (size_t)e * R * C;
    __bf16* oute = out + (size_t)e * R * C;
    int c0 = blockIdx.x * 32, r0 = blockIdx.y * 32;
    int tx = threadIdx.x, ty = threadIdx.y; // 32 x 8
#pragma unroll
    for (int j = 0; j < 4; ++j)
        t[ty + 8 * j][tx] = ine[(size_t)(r0 + ty + 8 * j) * C + c0 + tx];
    __syncthreads();
#pragma unroll
    for (int j = 0; j < 4; ++j)
        oute[(size_t)(c0 + ty + 8 * j) * R + r0 + tx] = (__bf16)t[tx][ty + 8 * j];
}

// ------------------------------------------------------------------ gating
__global__ void gate_kernel(const float* __restrict__ x,
                            const float* __restrict__ gw,
                            float* __restrict__ gates,
                            float* __restrict__ topv,
                            int*   __restrict__ topi) {
    __shared__ float gws[DIM * NEXP]; // 32 KB
    int tid = threadIdx.x;
    for (int i = tid; i < DIM * NEXP; i += 256) gws[i] = gw[i];
    __syncthreads();

    int wave = tid >> 5, lane = tid & 31;
    int n = blockIdx.x * 8 + wave;

    float acc[NEXP];
#pragma unroll
    for (int e = 0; e < NEXP; ++e) acc[e] = 0.f;

    const float* xr = x + (size_t)n * DIM;
    for (int d = lane; d < DIM; d += 32) {
        float xv = xr[d];
        const float* g = &gws[d * NEXP];
#pragma unroll
        for (int e = 0; e < NEXP; ++e) acc[e] += xv * g[e];
    }
#pragma unroll
    for (int off = 16; off > 0; off >>= 1) {
#pragma unroll
        for (int e = 0; e < NEXP; ++e) acc[e] += __shfl_xor(acc[e], off, 32);
    }

    if (lane == 0) {
        float mx = acc[0];
#pragma unroll
        for (int e = 1; e < NEXP; ++e) mx = fmaxf(mx, acc[e]);
        float s = 0.f, p[NEXP];
#pragma unroll
        for (int e = 0; e < NEXP; ++e) { p[e] = expf(acc[e] - mx); s += p[e]; }
        float inv = 1.f / s;
#pragma unroll
        for (int e = 0; e < NEXP; ++e) { p[e] *= inv; gates[(size_t)n * NEXP + e] = p[e]; }
        int i0 = 0; float v0 = p[0];
#pragma unroll
        for (int e = 1; e < NEXP; ++e) if (p[e] > v0) { v0 = p[e]; i0 = e; }
        int i1 = -1; float v1 = -1.f;
#pragma unroll
        for (int e = 0; e < NEXP; ++e) if (e != i0 && p[e] > v1) { v1 = p[e]; i1 = e; }
        topv[n * 2 + 0] = v0; topv[n * 2 + 1] = v1;
        topi[n * 2 + 0] = i0; topi[n * 2 + 1] = i1;
    }
}

// ---------------------------------------------------------------- routing
__global__ void route_kernel(const float* __restrict__ topv,
                             const int*   __restrict__ topi,
                             int*   __restrict__ combp,
                             float* __restrict__ combw,
                             int*   __restrict__ slot_token) {
    __shared__ float gl[256];
    __shared__ int   el[256];
    int tid = threadIdx.x;
    int gi  = blockIdx.x * 256 + tid;
    float g = topv[gi];
    int   e = topi[gi];
    int cnt = 0;
    for (int t = 0; t < NA / 256; ++t) {
        int j0 = t * 256;
        gl[tid] = topv[j0 + tid];
        el[tid] = topi[j0 + tid];
        __syncthreads();
#pragma unroll 8
        for (int j = 0; j < 256; ++j) {
            int jg = j0 + j;
            bool before = (gl[j] > g) || (gl[j] == g && jg < gi);
            cnt += (el[j] == e && before) ? 1 : 0;
        }
        __syncthreads();
    }
    bool keep = cnt < CAP;
    int posc = keep ? cnt : (CAP - 1);
    combp[gi] = posc;
    combw[gi] = keep ? g : 0.f;
    if (keep) slot_token[e * CAP + posc] = gi >> 1; // kept slots unique
}

// ------------------------------------------------------------- GEMM1 + GELU
// 128x128 tile, 256 thr (8 waves), K-step 32, double-buffered async LDS fills.
__global__ void gemm1_kernel(const __bf16* __restrict__ xb,
                             const __bf16* __restrict__ zrow,
                             const int*    __restrict__ slot_token,
                             const __bf16* __restrict__ w1t,
                             const float*  __restrict__ b1,
                             __bf16* __restrict__ hbuf,
                             int chunk) {
    __shared__ __bf16 As[2][128][40];   // [buf][m][k], 80B row stride (16B aligned)
    __shared__ __bf16 Bs[2][128][40];   // [buf][n][k]
    __shared__ int toks[128];

    int tid = threadIdx.x;
    int e   = blockIdx.z;
    int m0g = blockIdx.x * 128;                 // slot base
    int h0g = chunk * HC + blockIdx.y * 128;    // global hidden col base

    if (tid < 128) toks[tid] = slot_token[e * CAP + m0g + tid];
    __syncthreads();

    // per-thread fill assignment: 2 threads per row, 16 elems each
    int frow = tid >> 1, fc0 = (tid & 1) * 16;
    int tk = toks[frow];
    const __bf16* arow_src = ((tk >= 0) ? (xb + (size_t)tk * DIM) : zrow) + fc0;
    const __bf16* brow_src = w1t + (size_t)(h0g + frow) * DIM + fc0;
    unsigned a_lds[2] = { (unsigned)(uintptr_t)&As[0][frow][fc0],
                          (unsigned)(uintptr_t)&As[1][frow][fc0] };
    unsigned b_lds[2] = { (unsigned)(uintptr_t)&Bs[0][frow][fc0],
                          (unsigned)(uintptr_t)&Bs[1][frow][fc0] };

    auto issue = [&](int buf, int k0) {
        async_b128(a_lds[buf],      arow_src + k0);
        async_b128(a_lds[buf] + 16, arow_src + k0 + 8);
        async_b128(b_lds[buf],      brow_src + k0);
        async_b128(b_lds[buf] + 16, brow_src + k0 + 8);
    };

    v8f acc[8] = {};
    int wv = tid >> 5, lane = tid & 31, half = lane >> 4, mlo = lane & 15;

    issue(0, 0);
    for (int k0 = 0, it = 0; k0 < DIM; k0 += 32, ++it) {
        int cur = it & 1;
        __syncthreads();                       // prev readers of buf[1-cur] done
        bool has_next = (k0 + 32 < DIM);
        if (has_next) { issue(1 - cur, k0 + 32); wait_async4(); }
        else          { wait_async0(); }
        __syncthreads();                       // buf[cur] complete for all waves

        v16bf a;
        const __bf16* arow_lds = &As[cur][wv * 16 + mlo][0];
#pragma unroll
        for (int i = 0; i < 16; ++i) {
            int kk = half * 16 + i;
            a[i] = arow_lds[kk];
        }
#pragma unroll
        for (int nt = 0; nt < 8; ++nt) {
            v16bf b;
            const __bf16* brow = &Bs[cur][nt * 16 + mlo][half * 16];
#pragma unroll
            for (int i = 0; i < 16; ++i) b[i] = brow[i];
            acc[nt] = __builtin_amdgcn_wmma_f32_16x16x32_bf16(
                false, a, false, b, (short)0, acc[nt], false, false);
        }
    }

    const float* b1e = b1 + e * HID;
#pragma unroll
    for (int nt = 0; nt < 8; ++nt) {
#pragma unroll
        for (int r = 0; r < 8; ++r) {
            int row = wv * 16 + half * 8 + r;
            int hc  = blockIdx.y * 128 + nt * 16 + mlo;   // col within chunk
            float v = acc[nt][r] + b1e[chunk * HC + hc];
            v = 0.5f * v * (1.f + erff(v * 0.70710678118654752f));
            hbuf[((size_t)e * CAP + m0g + row) * HC + hc] = (__bf16)v;
        }
    }
}

// ------------------------------------------------------------------- GEMM2
__global__ void gemm2_kernel(const __bf16* __restrict__ hbuf,
                             const __bf16* __restrict__ w2t,
                             const float*  __restrict__ b2,
                             float* __restrict__ y,
                             int chunk) {
    __shared__ __bf16 As[2][128][40];
    __shared__ __bf16 Bs[2][128][40];

    int tid = threadIdx.x;
    int e   = blockIdx.z;
    int m0g = blockIdx.x * 128;
    int d0  = blockIdx.y * 128;

    int frow = tid >> 1, fc0 = (tid & 1) * 16;
    const __bf16* arow_src = hbuf + ((size_t)e * CAP + m0g + frow) * HC + fc0;
    const __bf16* brow_src = w2t + (size_t)(d0 + frow) * HID + chunk * HC + fc0;
    unsigned a_lds[2] = { (unsigned)(uintptr_t)&As[0][frow][fc0],
                          (unsigned)(uintptr_t)&As[1][frow][fc0] };
    unsigned b_lds[2] = { (unsigned)(uintptr_t)&Bs[0][frow][fc0],
                          (unsigned)(uintptr_t)&Bs[1][frow][fc0] };

    auto issue = [&](int buf, int k0) {
        async_b128(a_lds[buf],      arow_src + k0);
        async_b128(a_lds[buf] + 16, arow_src + k0 + 8);
        async_b128(b_lds[buf],      brow_src + k0);
        async_b128(b_lds[buf] + 16, brow_src + k0 + 8);
    };

    v8f acc[8] = {};
    int wv = tid >> 5, lane = tid & 31, half = lane >> 4, mlo = lane & 15;

    issue(0, 0);
    for (int k0 = 0, it = 0; k0 < HC; k0 += 32, ++it) {
        int cur = it & 1;
        __syncthreads();
        bool has_next = (k0 + 32 < HC);
        if (has_next) { issue(1 - cur, k0 + 32); wait_async4(); }
        else          { wait_async0(); }
        __syncthreads();

        v16bf a;
        const __bf16* arow_lds = &As[cur][wv * 16 + mlo][0];
#pragma unroll
        for (int i = 0; i < 16; ++i) {
            int kk = half * 16 + i;
            a[i] = arow_lds[kk];
        }
#pragma unroll
        for (int nt = 0; nt < 8; ++nt) {
            v16bf b;
            const __bf16* brow = &Bs[cur][nt * 16 + mlo][half * 16];
#pragma unroll
            for (int i = 0; i < 16; ++i) b[i] = brow[i];
            acc[nt] = __builtin_amdgcn_wmma_f32_16x16x32_bf16(
                false, a, false, b, (short)0, acc[nt], false, false);
        }
    }

    const float* b2e = b2 + e * DIM;
#pragma unroll
    for (int nt = 0; nt < 8; ++nt) {
#pragma unroll
        for (int r = 0; r < 8; ++r) {
            int row = wv * 16 + half * 8 + r;
            int col = d0 + nt * 16 + mlo;
            size_t yi = ((size_t)e * CAP + m0g + row) * DIM + col;
            float v = acc[nt][r];
            if (chunk == 0) y[yi] = v + b2e[col];
            else            y[yi] += v;   // single writer, stream-ordered -> deterministic
        }
    }
}

// ----------------------------------------------------------------- combine
__global__ void combine_kernel(const float* __restrict__ y,
                               const int*   __restrict__ topi,
                               const int*   __restrict__ combp,
                               const float* __restrict__ combw,
                               float* __restrict__ out) {
    int n = blockIdx.x;
    int e0 = topi[n * 2 + 0], e1 = topi[n * 2 + 1];
    int p0 = combp[n * 2 + 0], p1 = combp[n * 2 + 1];
    float w0 = combw[n * 2 + 0], w1 = combw[n * 2 + 1];
    const float* y0 = y + ((size_t)e0 * CAP + p0) * DIM;
    const float* y1 = y + ((size_t)e1 * CAP + p1) * DIM;
    float* o = out + (size_t)n * DIM;
    for (int d = threadIdx.x; d < DIM; d += 128)
        o[d] = w0 * y0[d] + w1 * y1[d];
}

// ---------------------------------------------------------------- aux loss
__global__ void aux_kernel(const float* __restrict__ gates,
                           const int*   __restrict__ topi,
                           float* __restrict__ out_last) {
    __shared__ float sme[256][NEXP];
    __shared__ float sce[256][NEXP];
    int tid = threadIdx.x;
    float me[NEXP], ce[NEXP];
#pragma unroll
    for (int e = 0; e < NEXP; ++e) { me[e] = 0.f; ce[e] = 0.f; }
    for (int n = tid; n < NTOK; n += 256) {
#pragma unroll
        for (int e = 0; e < NEXP; ++e) me[e] += gates[(size_t)n * NEXP + e];
        ce[topi[n * 2]] += 1.f;
    }
#pragma unroll
    for (int e = 0; e < NEXP; ++e) { sme[tid][e] = me[e]; sce[tid][e] = ce[e]; }
    __syncthreads();
    for (int s = 128; s > 0; s >>= 1) {
        if (tid < s) {
#pragma unroll
            for (int e = 0; e < NEXP; ++e) {
                sme[tid][e] += sme[tid + s][e];
                sce[tid][e] += sce[tid + s][e];
            }
        }
        __syncthreads();
    }
    if (tid == 0) {
        float l = 0.f;
#pragma unroll
        for (int e = 0; e < NEXP; ++e)
            l += (sme[0][e] / (float)NTOK) * (sce[0][e] / (float)NTOK);
        out_last[0] = l * (float)NEXP * 0.01f;
    }
}

// ------------------------------------------------------------------ driver
extern "C" void kernel_launch(void* const* d_in, const int* in_sizes, int n_in,
                              void* d_out, int out_size, void* d_ws, size_t ws_size,
                              hipStream_t stream) {
    (void)in_sizes; (void)n_in; (void)ws_size;
    const float* x   = (const float*)d_in[0];
    const float* gw  = (const float*)d_in[1];
    const float* w1  = (const float*)d_in[2];
    const float* b1  = (const float*)d_in[3];
    const float* w2  = (const float*)d_in[4];
    const float* b2  = (const float*)d_in[5];
    float* out = (float*)d_out;

    char* ws = (char*)d_ws;
    size_t off = 0;
    float* gates = (float*)(ws + off); off += (size_t)NTOK * NEXP * 4;     // 512 KB
    float* topv  = (float*)(ws + off); off += (size_t)NA * 4;
    int*   topi  = (int*)  (ws + off); off += (size_t)NA * 4;
    int*   combp = (int*)  (ws + off); off += (size_t)NA * 4;
    float* combw = (float*)(ws + off); off += (size_t)NA * 4;
    int*   slot  = (int*)  (ws + off); off += (size_t)NEXP * CAP * 4;      // 160 KB
    __bf16* zrow = (__bf16*)(ws + off); off += (size_t)DIM * 2;            // 2 KB
    off = (off + 255) & ~(size_t)255;
    __bf16* xb   = (__bf16*)(ws + off); off += (size_t)NTOK * DIM * 2;       // 33.5 MB
    __bf16* w1t  = (__bf16*)(ws + off); off += (size_t)NEXP * DIM * HID * 2; // 67 MB
    __bf16* w2t  = (__bf16*)(ws + off); off += (size_t)NEXP * HID * DIM * 2; // 67 MB
    __bf16* hbuf = (__bf16*)(ws + off); off += (size_t)NEXP * CAP * HC * 2;  // 42 MB
    off = (off + 255) & ~(size_t)255;
    float* y     = (float*)(ws + off); off += (size_t)NEXP * CAP * DIM * 4;  // 160 MB

    init_kernel<<<(NEXP * CAP + 255) / 256, 256, 0, stream>>>(slot, zrow);
    cvt_x_kernel<<<(NTOK * DIM) / 1024, 256, 0, stream>>>(x, xb);
    transpose_cvt_kernel<<<dim3(HID / 32, DIM / 32, NEXP), dim3(32, 8), 0, stream>>>(w1, w1t, DIM, HID);
    transpose_cvt_kernel<<<dim3(DIM / 32, HID / 32, NEXP), dim3(32, 8), 0, stream>>>(w2, w2t, HID, DIM);

    gate_kernel<<<NTOK / 8, 256, 0, stream>>>(x, gw, gates, topv, topi);
    route_kernel<<<NA / 256, 256, 0, stream>>>(topv, topi, combp, combw, slot);

    for (int c = 0; c < NCHUNK; ++c) {
        gemm1_kernel<<<dim3(CAP / 128, HC / 128, NEXP), 256, 0, stream>>>(xb, zrow, slot, w1t, b1, hbuf, c);
        gemm2_kernel<<<dim3(CAP / 128, DIM / 128, NEXP), 256, 0, stream>>>(hbuf, w2t, b2, y, c);
    }

    combine_kernel<<<NTOK, 128, 0, stream>>>(y, topi, combp, combw, out);
    aux_kernel<<<1, 256, 0, stream>>>(gates, topi, out + ((size_t)out_size - 1));
}
